// NanoLCM_9457517985937
// MI455X (gfx1250) — compile-verified, matrix operand order checked by reading, not run on hardware
//
#include <hip/hip_runtime.h>
#include <math.h>

#define BTOT 16384

typedef __attribute__((ext_vector_type(16))) _Float16 v16h;
typedef __attribute__((ext_vector_type(8)))  float    v8f;

__device__ __forceinline__ float sx(float v, int m) { return __shfl_xor(v, m, 32); }

__device__ __forceinline__ float wsum32(float v) {
#pragma unroll
  for (int m = 1; m < 32; m <<= 1) v += __shfl_xor(v, m, 32);
  return v;
}

__device__ __forceinline__ unsigned pack2(float a, float b) {
  union { _Float16 h[2]; unsigned u; } p;
  p.h[0] = (_Float16)a; p.h[1] = (_Float16)b;
  return p.u;
}

// ---- WMMA fragment loaders (wave32 layouts per CDNA5 ISA 7.12.2) ----
// A: 16x32 f16 row-major in LDS, stride lda (even). lane holds row m=lane&15;
// lanes 0-15 cover K 0-7 & 16-23, lanes 16-31 cover K 8-15 & 24-31.
// k-pairs are contiguous + dword-aligned -> ds_load_(2addr_)b32, no packing movs.
__device__ __forceinline__ v16h ldfragA32(const _Float16* A, int lda, int lane) {
  const unsigned* A32 = (const unsigned*)A;  // caller guarantees even-element, 4B-aligned base
  int m = lane & 15, kb2 = (lane >> 4) << 2, lda2 = lda >> 1;
  union { unsigned u[8]; v16h h; } t;
#pragma unroll
  for (int d = 0; d < 8; ++d) {
    int k2 = ((d & 4) << 1) + kb2 + (d & 3);  // element-pair index = k/2
    t.u[d] = A32[m * lda2 + k2];
  }
  return t.h;
}
// B: 32x16 slice of a K x 96 weight tile stored PACKED as
// Wp[(k/2)*96 + n] = {W[k][n](lo), W[k+1][n](hi)} -> 8x ds_load_b32.
__device__ __forceinline__ v16h ldfragBp(const unsigned* Wp, int lane) {
  int n = lane & 15, kb2 = (lane >> 4) << 2;
  union { unsigned u[8]; v16h h; } t;
#pragma unroll
  for (int d = 0; d < 8; ++d) {
    int k2 = ((d & 4) << 1) + kb2 + (d & 3);
    t.u[d] = Wp[k2 * 96 + n];
  }
  return t.h;
}

// =====================================================================
// Kernel 1: quantum circuit + hpp + scale/shift modulation (wave per row)
// State: 64 complex amps; lane holds indices lane (slot0) and lane+32 (slot1).
// qubit q <-> bit (5-q) of the amplitude index.
// =====================================================================
__device__ __forceinline__ void run_circuit(const float* __restrict__ ang,
                                            const float* __restrict__ th,
                                            int lane, float* meas) {
  float r0 = (lane == 0) ? 1.f : 0.f, i0 = 0.f, r1 = 0.f, i1 = 0.f;

  auto RX = [&](int bit, float t) {
    float c = cosf(0.5f * t), s = sinf(0.5f * t);
    if (bit == 5) {
      float nr0 = c * r0 + s * i1, ni0 = c * i0 - s * r1;
      float nr1 = c * r1 + s * i0, ni1 = c * i1 - s * r0;
      r0 = nr0; i0 = ni0; r1 = nr1; i1 = ni1;
    } else {
      int m = 1 << bit;
      float pr0 = sx(r0, m), pi0 = sx(i0, m), pr1 = sx(r1, m), pi1 = sx(i1, m);
      r0 = c * r0 + s * pi0; i0 = c * i0 - s * pr0;
      r1 = c * r1 + s * pi1; i1 = c * i1 - s * pr1;
    }
  };
  auto RY = [&](int bit, float t) {
    float c = cosf(0.5f * t), s = sinf(0.5f * t);
    if (bit == 5) {
      float nr0 = c * r0 - s * r1, ni0 = c * i0 - s * i1;
      float nr1 = c * r1 + s * r0, ni1 = c * i1 + s * i0;
      r0 = nr0; i0 = ni0; r1 = nr1; i1 = ni1;
    } else {
      int m = 1 << bit;
      float sg  = ((lane >> bit) & 1) ? s : -s;
      float pr0 = sx(r0, m), pi0 = sx(i0, m), pr1 = sx(r1, m), pi1 = sx(i1, m);
      r0 = c * r0 + sg * pr0; i0 = c * i0 + sg * pi0;
      r1 = c * r1 + sg * pr1; i1 = c * i1 + sg * pi1;
    }
  };
  auto RZ = [&](int bit, float t) {
    float c = cosf(0.5f * t), s = sinf(0.5f * t);
    if (bit == 5) {
      float nr0 = r0 * c + i0 * s, ni0 = i0 * c - r0 * s; // e^{-it/2}
      float nr1 = r1 * c - i1 * s, ni1 = i1 * c + r1 * s; // e^{+it/2}
      r0 = nr0; i0 = ni0; r1 = nr1; i1 = ni1;
    } else {
      float e   = ((lane >> bit) & 1) ? s : -s;
      float nr0 = r0 * c - i0 * e, ni0 = i0 * c + r0 * e;
      float nr1 = r1 * c - i1 * e, ni1 = i1 * c + r1 * e;
      r0 = nr0; i0 = ni0; r1 = nr1; i1 = ni1;
    }
  };
  auto CNOT = [&](int cq, int tq) {
    int cbit = 5 - cq, tbit = 5 - tq;
    if (tbit == 5) {                     // target is slot bit
      if ((lane >> cbit) & 1) {
        float t = r0; r0 = r1; r1 = t;
        t = i0; i0 = i1; i1 = t;
      }
    } else if (cbit == 5) {              // control is slot bit: slot1 swaps target
      int m = 1 << tbit;
      r1 = sx(r1, m); i1 = sx(i1, m);
    } else {
      int m  = 1 << tbit;
      int ct = (lane >> cbit) & 1;
      float pr0 = sx(r0, m), pi0 = sx(i0, m), pr1 = sx(r1, m), pi1 = sx(i1, m);
      if (ct) { r0 = pr0; i0 = pi0; r1 = pr1; i1 = pi1; }
    }
  };

#pragma unroll
  for (int q = 0; q < 6; ++q) { RX(5 - q, ang[q]); RZ(5 - q, 0.5f * ang[q]); }
  for (int l = 0; l < 3; ++l) {
#pragma unroll
    for (int q = 0; q < 5; ++q) CNOT(q, q + 1);
    CNOT(5, 0);
#pragma unroll
    for (int q = 0; q < 6; ++q) {
      RX(5 - q, th[(l * 6 + q) * 3 + 0]);
      RY(5 - q, th[(l * 6 + q) * 3 + 1]);
      RZ(5 - q, th[(l * 6 + q) * 3 + 2]);
    }
  }
#pragma unroll
  for (int k = 0; k < 8; ++k) {
    int q = k % 6, bit = 5 - q;
    if ((k & 1) == 0) {                  // <Z_q> = 1 - 2*P(bit=1)
      float loc;
      if (bit == 5) loc = r1 * r1 + i1 * i1;
      else loc = ((lane >> bit) & 1) ? (r0 * r0 + i0 * i0 + r1 * r1 + i1 * i1) : 0.f;
      meas[k] = 1.f - 2.f * wsum32(loc);
    } else {                             // Re<psi|X_q|psi> ; bit<5 for odd k
      int m = 1 << bit;
      float loc = r0 * sx(r0, m) + i0 * sx(i0, m) + r1 * sx(r1, m) + i1 * sx(i1, m);
      meas[k] = wsum32(loc);
    }
  }
}

__global__ __launch_bounds__(256) void k_quantum_mod(
    const float* __restrict__ x,
    const float* __restrict__ th0, const float* __restrict__ th1,
    const float* __restrict__ hpw, const float* __restrict__ hpb,
    const float* __restrict__ hng, const float* __restrict__ hnb,
    const float* __restrict__ sw1, const float* __restrict__ sb1,
    const float* __restrict__ sw2, const float* __restrict__ sb2,
    const float* __restrict__ tw1, const float* __restrict__ tb1,
    const float* __restrict__ tw2, const float* __restrict__ tb2,
    float* __restrict__ out,            // [0,B): logits (kernel2), [B,17B): h_ctx
    float* __restrict__ gmod, float* __restrict__ tmod) {
  __shared__ float sh_h[8][16];
  __shared__ float sh_u[8][100];
  const int lane = threadIdx.x & 31;
  const int wave = threadIdx.x >> 5;
  const int b = blockIdx.x * 8 + wave;

  // pool6 -> angles (6 groups of 128 contiguous cols)
  float ang[6];
#pragma unroll
  for (int g = 0; g < 6; ++g) {
    float s = 0.f;
#pragma unroll
    for (int j = 0; j < 4; ++j) s += x[(size_t)b * 768 + g * 128 + lane + 32 * j];
    s = wsum32(s) * (1.f / 128.f);
    s = fminf(fmaxf(s, -6.f), 6.f);
    ang[g] = s * 0.52359877559829887f;  // pi/6
  }

  float emb[16];
  run_circuit(ang, th0, lane, emb);
  run_circuit(ang, th1, lane, emb + 8);

  // hpp: tanh(proj) then LN over 16 (lanes 0..15 carry the 16 channels)
  {
    int j = lane & 15;
    float acc = hpb[j];
#pragma unroll
    for (int t = 0; t < 16; ++t) acc += hpw[j * 16 + t] * emb[t];
    float hv = tanhf(acc);
    float s1 = hv;
#pragma unroll
    for (int m = 1; m < 16; m <<= 1) s1 += __shfl_xor(s1, m, 32);
    float mean = s1 * (1.f / 16.f);
    float d = hv - mean, vv = d * d;
#pragma unroll
    for (int m = 1; m < 16; m <<= 1) vv += __shfl_xor(vv, m, 32);
    float hn = d * rsqrtf(vv * (1.f / 16.f) + 1e-5f) * hng[j] + hnb[j];
    if (lane < 16) {
      sh_h[wave][j] = hn;
      out[(size_t)BTOT + (size_t)b * 16 + j] = hn;
    }
  }
  __syncthreads();

  // scale path: s = W2 @ silu(W1 @ h + b1) + b2 ; store 1+tanh(s)
#pragma unroll
  for (int j = 0; j < 3; ++j) {
    int o = lane + 32 * j;
    float a = sb1[o];
#pragma unroll
    for (int t = 0; t < 16; ++t) a += sw1[o * 16 + t] * sh_h[wave][t];
    sh_u[wave][o] = a / (1.f + expf(-a));
  }
  __syncthreads();
#pragma unroll
  for (int j = 0; j < 3; ++j) {
    int o = lane + 32 * j;
    float a = sb2[o];
    for (int t = 0; t < 96; ++t) a += sw2[o * 96 + t] * sh_u[wave][t];
    gmod[(size_t)b * 96 + o] = 1.f + tanhf(a);
  }
  __syncthreads();
  // shift path
#pragma unroll
  for (int j = 0; j < 3; ++j) {
    int o = lane + 32 * j;
    float a = tb1[o];
#pragma unroll
    for (int t = 0; t < 16; ++t) a += tw1[o * 16 + t] * sh_h[wave][t];
    sh_u[wave][o] = a / (1.f + expf(-a));
  }
  __syncthreads();
#pragma unroll
  for (int j = 0; j < 3; ++j) {
    int o = lane + 32 * j;
    float a = tb2[o];
    for (int t = 0; t < 96; ++t) a += tw2[o * 96 + t] * sh_u[wave][t];
    tmod[(size_t)b * 96 + o] = a;
  }
}

// =====================================================================
// Kernel 2: fused transformer over 8 batch rows (64 token-rows) per WG.
// GEMMs use v_wmma_f32_16x16x32_f16 (f16 operands in LDS, f32 acc).
// =====================================================================
__global__ __launch_bounds__(256) void k_transformer(
    const float* __restrict__ x,
    const float* __restrict__ embw, const float* __restrict__ embb,
    const float* __restrict__ l1g, const float* __restrict__ l1b,
    const float* __restrict__ aiw, const float* __restrict__ aib,
    const float* __restrict__ aow, const float* __restrict__ aob,
    const float* __restrict__ l2g, const float* __restrict__ l2b,
    const float* __restrict__ m1w, const float* __restrict__ m1b,
    const float* __restrict__ m2w, const float* __restrict__ m2b,
    const float* __restrict__ hlg, const float* __restrict__ hlb,
    const float* __restrict__ hw, const float* __restrict__ hb,
    const float* __restrict__ gmod, const float* __restrict__ tmod,
    float* __restrict__ out) {
  extern __shared__ char smem[];
  float*    tok = (float*)smem;               // 64 x 96   f32 residual (24576 B)
  _Float16* qkv = (_Float16*)(tok + 64 * 96); // 64 x 288  f16 (36864 B)
  _Float16* Xh  = qkv + 64 * 288;             // 64 x 96   f16 (12288 B, A staging)
  _Float16* Hh  = Xh + 64 * 96;               // 64 x 192  f16 (24576 B, mlp hidden)
  unsigned* Wp  = (unsigned*)(Hh + 64 * 192); // 48 x 96   packed f16 pairs (18432 B)
  float*    gm  = (float*)(Wp + 48 * 96);     // 8 x 96
  float*    tm  = gm + 8 * 96;                // 8 x 96

  const int tid = threadIdx.x, lane = tid & 31, wave = tid >> 5;
  const int b0 = blockIdx.x * 8;

  // warm L2 for the block-invariant weights (global_prefetch_b8 path)
  if (tid == 0) {
    __builtin_prefetch(embw, 0, 3);
    __builtin_prefetch(aiw, 0, 3);
    __builtin_prefetch(aow, 0, 3);
    __builtin_prefetch(m1w, 0, 3);
    __builtin_prefetch(m2w, 0, 3);
  }

  for (int i = tid; i < 8 * 96; i += 256) {
    gm[i] = gmod[(size_t)b0 * 96 + i];
    tm[i] = tmod[(size_t)b0 * 96 + i];
  }
  // A := x chunks (row r = b_local*8+tok, k<96 ; CHUNK==96); packed dword stores
  {
    unsigned* X32 = (unsigned*)Xh;
    for (int i = tid; i < 64 * 48; i += 256) {
      int e = 2 * i, r = e / 96, k = e - 96 * r, bl = r >> 3, tk = r & 7;
      const float* src = x + (size_t)(b0 + bl) * 768 + tk * 96 + k;
      X32[i] = pack2(src[0], src[1]);
    }
  }
  // B := embed_w^T, packed pairs along K
  for (int i = tid; i < 48 * 96; i += 256) {
    int k2 = i / 96, n = i - 96 * k2;
    Wp[i] = pack2(embw[n * 96 + 2 * k2], embw[n * 96 + 2 * k2 + 1]);
  }
  __syncthreads();

  // ---- embed GEMM + bias + modulation -> tok
  for (int j = wave; j < 24; j += 8) {
    int mt = j / 6, nt = j - 6 * mt;
    v8f acc = {0.f, 0.f, 0.f, 0.f, 0.f, 0.f, 0.f, 0.f};
#pragma unroll
    for (int kt = 0; kt < 3; ++kt) {
      v16h a  = ldfragA32(Xh + mt * 16 * 96 + kt * 32, 96, lane);
      v16h bf = ldfragBp(Wp + kt * 16 * 96 + nt * 16, lane);
      acc = __builtin_amdgcn_wmma_f32_16x16x32_f16(false, a, false, bf, (short)0, acc, false, false);
    }
    int n = nt * 16 + (lane & 15);
    int mb = mt * 16 + ((lane >> 4) << 3);   // multiple of 8: all 8 rows in one batch elem
    int bl = mb >> 3;
    float bias = embb[n];
    float gmv = gm[bl * 96 + n], tmv = tm[bl * 96 + n];  // hoisted, loop-invariant
#pragma unroll
    for (int r = 0; r < 8; ++r)
      tok[(mb + r) * 96 + n] = (acc[r] + bias) * gmv + tmv;
  }
  __syncthreads();

  // ---- LN1 -> Xh
  for (int rr = 0; rr < 8; ++rr) {
    int row = wave * 8 + rr;
    float v0 = tok[row * 96 + lane], v1 = tok[row * 96 + lane + 32], v2 = tok[row * 96 + lane + 64];
    float mean = wsum32(v0 + v1 + v2) * (1.f / 96.f);
    float d0 = v0 - mean, d1 = v1 - mean, d2 = v2 - mean;
    float rs = rsqrtf(wsum32(d0 * d0 + d1 * d1 + d2 * d2) * (1.f / 96.f) + 1e-5f);
    Xh[row * 96 + lane]      = (_Float16)(d0 * rs * l1g[lane] + l1b[lane]);
    Xh[row * 96 + lane + 32] = (_Float16)(d1 * rs * l1g[lane + 32] + l1b[lane + 32]);
    Xh[row * 96 + lane + 64] = (_Float16)(d2 * rs * l1g[lane + 64] + l1b[lane + 64]);
  }
  __syncthreads();

  // ---- qkv GEMM (N=288 in 3 chunks of 96)
  for (int c = 0; c < 3; ++c) {
    for (int i = tid; i < 48 * 96; i += 256) {
      int k2 = i / 96, n = i - 96 * k2;
      const float* w = aiw + (size_t)(c * 96 + n) * 96 + 2 * k2;
      Wp[i] = pack2(w[0], w[1]);
    }
    __syncthreads();
    for (int j = wave; j < 24; j += 8) {
      int mt = j / 6, nt = j - 6 * mt;
      v8f acc = {0.f, 0.f, 0.f, 0.f, 0.f, 0.f, 0.f, 0.f};
#pragma unroll
      for (int kt = 0; kt < 3; ++kt) {
        v16h a  = ldfragA32(Xh + mt * 16 * 96 + kt * 32, 96, lane);
        v16h bf = ldfragBp(Wp + kt * 16 * 96 + nt * 16, lane);
        acc = __builtin_amdgcn_wmma_f32_16x16x32_f16(false, a, false, bf, (short)0, acc, false, false);
      }
      int n = nt * 16 + (lane & 15);
      int mb = mt * 16 + ((lane >> 4) << 3);
      float bias = aib[c * 96 + n];
#pragma unroll
      for (int r = 0; r < 8; ++r)
        qkv[(mb + r) * 288 + c * 96 + n] = (_Float16)(acc[r] + bias);
    }
    __syncthreads();
  }

  // ---- attention: 256 threads = 8 batch x 4 heads x 8 query tokens
  {
    int bl = tid >> 5, hh = (tid >> 3) & 3, qt = tid & 7;
    int rq = bl * 8 + qt;
    float qv[24];
#pragma unroll
    for (int d = 0; d < 24; ++d) qv[d] = (float)qkv[rq * 288 + hh * 24 + d];
    float sc[8], mx = -1e30f;
#pragma unroll
    for (int kt = 0; kt < 8; ++kt) {
      float s = 0.f;
#pragma unroll
      for (int d = 0; d < 24; ++d) s += qv[d] * (float)qkv[(bl * 8 + kt) * 288 + 96 + hh * 24 + d];
      sc[kt] = s * 0.20412414523193153f;  // 1/sqrt(24)
      mx = fmaxf(mx, sc[kt]);
    }
    float sum = 0.f;
#pragma unroll
    for (int kt = 0; kt < 8; ++kt) { sc[kt] = expf(sc[kt] - mx); sum += sc[kt]; }
    float inv = 1.f / sum;
    float o[24];
#pragma unroll
    for (int d = 0; d < 24; ++d) o[d] = 0.f;
#pragma unroll
    for (int kt = 0; kt < 8; ++kt) {
      float w = sc[kt] * inv;
#pragma unroll
      for (int d = 0; d < 24; ++d) o[d] += w * (float)qkv[(bl * 8 + kt) * 288 + 192 + hh * 24 + d];
    }
#pragma unroll
    for (int d = 0; d < 24; ++d) Xh[rq * 96 + hh * 24 + d] = (_Float16)o[d];
  }
  __syncthreads();

  // ---- attn_out GEMM: tok += a @ W^T + b
  for (int i = tid; i < 48 * 96; i += 256) {
    int k2 = i / 96, n = i - 96 * k2;
    Wp[i] = pack2(aow[n * 96 + 2 * k2], aow[n * 96 + 2 * k2 + 1]);
  }
  __syncthreads();
  for (int j = wave; j < 24; j += 8) {
    int mt = j / 6, nt = j - 6 * mt;
    v8f acc = {0.f, 0.f, 0.f, 0.f, 0.f, 0.f, 0.f, 0.f};
#pragma unroll
    for (int kt = 0; kt < 3; ++kt) {
      v16h a  = ldfragA32(Xh + mt * 16 * 96 + kt * 32, 96, lane);
      v16h bf = ldfragBp(Wp + kt * 16 * 96 + nt * 16, lane);
      acc = __builtin_amdgcn_wmma_f32_16x16x32_f16(false, a, false, bf, (short)0, acc, false, false);
    }
    int n = nt * 16 + (lane & 15);
    int mb = mt * 16 + ((lane >> 4) << 3);
    float bias = aob[n];
    float old[8];
#pragma unroll
    for (int r = 0; r < 8; ++r) old[r] = tok[(mb + r) * 96 + n];  // batched loads, one wait
#pragma unroll
    for (int r = 0; r < 8; ++r) tok[(mb + r) * 96 + n] = old[r] + acc[r] + bias;
  }
  __syncthreads();

  // ---- LN2 -> Xh
  for (int rr = 0; rr < 8; ++rr) {
    int row = wave * 8 + rr;
    float v0 = tok[row * 96 + lane], v1 = tok[row * 96 + lane + 32], v2 = tok[row * 96 + lane + 64];
    float mean = wsum32(v0 + v1 + v2) * (1.f / 96.f);
    float d0 = v0 - mean, d1 = v1 - mean, d2 = v2 - mean;
    float rs = rsqrtf(wsum32(d0 * d0 + d1 * d1 + d2 * d2) * (1.f / 96.f) + 1e-5f);
    Xh[row * 96 + lane]      = (_Float16)(d0 * rs * l2g[lane] + l2b[lane]);
    Xh[row * 96 + lane + 32] = (_Float16)(d1 * rs * l2g[lane + 32] + l2b[lane + 32]);
    Xh[row * 96 + lane + 64] = (_Float16)(d2 * rs * l2g[lane + 64] + l2b[lane + 64]);
  }
  __syncthreads();

  // ---- MLP1: h = gelu(xn2 @ W1^T + b1), N=192 in 2 chunks
  for (int c = 0; c < 2; ++c) {
    for (int i = tid; i < 48 * 96; i += 256) {
      int k2 = i / 96, n = i - 96 * k2;
      const float* w = m1w + (size_t)(c * 96 + n) * 96 + 2 * k2;
      Wp[i] = pack2(w[0], w[1]);
    }
    __syncthreads();
    for (int j = wave; j < 24; j += 8) {
      int mt = j / 6, nt = j - 6 * mt;
      v8f acc = {0.f, 0.f, 0.f, 0.f, 0.f, 0.f, 0.f, 0.f};
#pragma unroll
      for (int kt = 0; kt < 3; ++kt) {
        v16h a  = ldfragA32(Xh + mt * 16 * 96 + kt * 32, 96, lane);
        v16h bf = ldfragBp(Wp + kt * 16 * 96 + nt * 16, lane);
        acc = __builtin_amdgcn_wmma_f32_16x16x32_f16(false, a, false, bf, (short)0, acc, false, false);
      }
      int n = nt * 16 + (lane & 15);
      int mb = mt * 16 + ((lane >> 4) << 3);
      float bias = m1b[c * 96 + n];
#pragma unroll
      for (int r = 0; r < 8; ++r) {
        float v = acc[r] + bias;
        v = 0.5f * v * (1.f + erff(v * 0.70710678118654752f));  // exact gelu
        Hh[(mb + r) * 192 + c * 96 + n] = (_Float16)v;
      }
    }
    __syncthreads();
  }

  // ---- MLP2: tok += h @ W2^T + b2 (K=192, staged in 2 K-chunks)
  {
    v8f acc3[3];
#pragma unroll
    for (int jj = 0; jj < 3; ++jj)
      acc3[jj] = (v8f){0.f, 0.f, 0.f, 0.f, 0.f, 0.f, 0.f, 0.f};
    for (int kc = 0; kc < 2; ++kc) {
      for (int i = tid; i < 48 * 96; i += 256) {
        int k2 = i / 96, n = i - 96 * k2;
        const float* w = m2w + (size_t)n * 192 + kc * 96 + 2 * k2;
        Wp[i] = pack2(w[0], w[1]);
      }
      __syncthreads();
#pragma unroll
      for (int jj = 0; jj < 3; ++jj) {
        int j = wave + jj * 8;
        int mt = j / 6, nt = j - 6 * mt;
#pragma unroll
        for (int kt = 0; kt < 3; ++kt) {
          v16h a  = ldfragA32(Hh + mt * 16 * 192 + kc * 96 + kt * 32, 192, lane);
          v16h bf = ldfragBp(Wp + kt * 16 * 96 + nt * 16, lane);
          acc3[jj] = __builtin_amdgcn_wmma_f32_16x16x32_f16(false, a, false, bf, (short)0, acc3[jj], false, false);
        }
      }
      __syncthreads();
    }
#pragma unroll
    for (int jj = 0; jj < 3; ++jj) {
      int j = wave + jj * 8;
      int mt = j / 6, nt = j - 6 * mt;
      int n = nt * 16 + (lane & 15);
      int mb = mt * 16 + ((lane >> 4) << 3);
      float bias = m2b[n];
      float old[8];
#pragma unroll
      for (int r = 0; r < 8; ++r) old[r] = tok[(mb + r) * 96 + n];
#pragma unroll
      for (int r = 0; r < 8; ++r) tok[(mb + r) * 96 + n] = old[r] + acc3[jj][r] + bias;
    }
  }
  __syncthreads();

  // ---- pooled mean over tokens, head LN, logit (wave w -> batch w)
  {
    int bl = wave;
    float p0 = 0.f, p1 = 0.f, p2 = 0.f;
#pragma unroll
    for (int t = 0; t < 8; ++t) {
      int row = bl * 8 + t;
      p0 += tok[row * 96 + lane];
      p1 += tok[row * 96 + lane + 32];
      p2 += tok[row * 96 + lane + 64];
    }
    p0 *= 0.125f; p1 *= 0.125f; p2 *= 0.125f;
    float mean = wsum32(p0 + p1 + p2) * (1.f / 96.f);
    float d0 = p0 - mean, d1 = p1 - mean, d2 = p2 - mean;
    float rs = rsqrtf(wsum32(d0 * d0 + d1 * d1 + d2 * d2) * (1.f / 96.f) + 1e-5f);
    float lg = (d0 * rs * hlg[lane] + hlb[lane]) * hw[lane]
             + (d1 * rs * hlg[lane + 32] + hlb[lane + 32]) * hw[lane + 32]
             + (d2 * rs * hlg[lane + 64] + hlb[lane + 64]) * hw[lane + 64];
    lg = wsum32(lg);
    if (lane == 0) out[b0 + bl] = lg + hb[0];
  }
}

extern "C" void kernel_launch(void* const* d_in, const int* in_sizes, int n_in,
                              void* d_out, int out_size, void* d_ws, size_t ws_size,
                              hipStream_t stream) {
  (void)in_sizes; (void)n_in; (void)out_size; (void)ws_size;
  const float* x    = (const float*)d_in[0];
  const float* th0  = (const float*)d_in[1];
  const float* th1  = (const float*)d_in[2];
  const float* hpw  = (const float*)d_in[3];
  const float* hpb  = (const float*)d_in[4];
  const float* hng  = (const float*)d_in[5];
  const float* hnb  = (const float*)d_in[6];
  const float* sw1  = (const float*)d_in[7];
  const float* sb1  = (const float*)d_in[8];
  const float* sw2  = (const float*)d_in[9];
  const float* sb2  = (const float*)d_in[10];
  const float* tw1  = (const float*)d_in[11];
  const float* tb1  = (const float*)d_in[12];
  const float* tw2  = (const float*)d_in[13];
  const float* tb2  = (const float*)d_in[14];
  const float* embw = (const float*)d_in[15];
  const float* embb = (const float*)d_in[16];
  const float* l1g  = (const float*)d_in[17];
  const float* l1b  = (const float*)d_in[18];
  const float* aiw  = (const float*)d_in[19];
  const float* aib  = (const float*)d_in[20];
  const float* aow  = (const float*)d_in[21];
  const float* aob  = (const float*)d_in[22];
  const float* l2g  = (const float*)d_in[23];
  const float* l2b  = (const float*)d_in[24];
  const float* m1w  = (const float*)d_in[25];
  const float* m1b  = (const float*)d_in[26];
  const float* m2w  = (const float*)d_in[27];
  const float* m2b  = (const float*)d_in[28];
  const float* hlg  = (const float*)d_in[29];
  const float* hlb  = (const float*)d_in[30];
  const float* hwv  = (const float*)d_in[31];
  const float* hbv  = (const float*)d_in[32];

  float* out  = (float*)d_out;
  float* gmod = (float*)d_ws;
  float* tmod = gmod + (size_t)BTOT * 96;

  dim3 blk(256);
  k_quantum_mod<<<BTOT / 8, blk, 0, stream>>>(
      x, th0, th1, hpw, hpb, hng, hnb,
      sw1, sb1, sw2, sb2, tw1, tb1, tw2, tb2,
      out, gmod, tmod);

  const size_t shbytes = 64 * 96 * 4      // tok f32
                       + 64 * 288 * 2     // qkv f16
                       + 64 * 96 * 2      // Xh f16
                       + 64 * 192 * 2     // Hh f16
                       + 48 * 96 * 4      // Wp packed f16 pairs
                       + 2 * 8 * 96 * 4;  // gm/tm f32  == 122880 B
  hipFuncSetAttribute(reinterpret_cast<const void*>(k_transformer),
                      hipFuncAttributeMaxDynamicSharedMemorySize, (int)shbytes);
  k_transformer<<<BTOT / 8, blk, shbytes, stream>>>(
      x, embw, embb, l1g, l1b, aiw, aib, aow, aob, l2g, l2b,
      m1w, m1b, m2w, m2b, hlg, hlb, hwv, hbv, gmod, tmod, out);
}
